// PiAwareAttention_86784109183604
// MI455X (gfx1250) — compile-verified, matrix operand order checked by reading, not run on hardware
//
#include <hip/hip_runtime.h>

// ---------------------------------------------------------------------------
// PiAwareAttention for MI455X (gfx1250, wave32, WMMA + Tensor Data Mover).
// Spectral FFT filter is replaced by its equivalent symmetric circulant matrix
// C (built on device from spectral_alpha), turning the whole model into
// bf16 WMMA GEMMs + a flash-attention kernel (f32 accumulate throughout).
// All GEMM tiles are staged with TDM (tensor_load_to_lds, TENSORcnt double-
// buffered, one barrier per K-step). pi_bias is pre-converted to bf16 to
// halve the dominant HBM stream.
// ---------------------------------------------------------------------------

typedef __attribute__((ext_vector_type(16))) __bf16 v16bf;
typedef __attribute__((ext_vector_type(8)))  float  v8f;
typedef __attribute__((ext_vector_type(4)))  unsigned int u32x4;
typedef __attribute__((ext_vector_type(8)))  unsigned int u32x8;
typedef unsigned short u16;
typedef long long      i64;
typedef unsigned long long u64;

struct alignas(16) U4 { unsigned int x, y, z, w; };
union Frag16 { v16bf v; U4 q[2]; };

static __device__ __forceinline__ u16 f2bf(float f) {
  union { float f; unsigned int u; } x; x.f = f;
  unsigned int r = 0x7FFFu + ((x.u >> 16) & 1u);   // round-to-nearest-even
  return (u16)((x.u + r) >> 16);
}

static __device__ __forceinline__ float bf2f(u16 v) {
  union { unsigned int u; float f; } x;
  x.u = ((unsigned int)v) << 16;
  return x.f;
}

static __device__ __forceinline__ v8f vzero8() {
  v8f z;
#pragma unroll
  for (int i = 0; i < 8; ++i) z[i] = 0.f;
  return z;
}

// Load one 16-bit A/B-style WMMA fragment from a row of bf16 data.
// Per the CDNA5 ISA 16-bit layout, lane-half 0 holds K {kb+0..7, kb+16..23},
// lane-half 1 holds K {kb+8..15, kb+24..31}.
static __device__ __forceinline__ v16bf load_frag(const u16* rowptr, int kbase, int half) {
  Frag16 f;
  f.q[0] = *(const U4*)(rowptr + kbase + half * 8);
  f.q[1] = *(const U4*)(rowptr + kbase + 16 + half * 8);
  return f.v;
}

static __device__ __forceinline__ v8f wmma_bf16(v16bf a, v16bf b, v8f c) {
  return __builtin_amdgcn_wmma_f32_16x16x32_bf16(false, a, false, b, (short)0, c,
                                                 false, false);
}

// ---------------------------------------------------------------------------
// Tensor Data Mover: async-copy a 2D bf16 tile [rows x w_elems] (global row
// stride = stride_elems) into LDS, inserting pad_amt+1 dwords of padding after
// every 2^(pad_int+1) dwords so LDS rows keep a bank-conflict-free stride.
// One wave instruction; completion tracked by TENSORcnt.
// D# per CDNA5 ISA 8.3/8.4: group0 {count=1, lds_addr, global_addr, type=2},
// group1 {data_size=2B, pad ctl, tensor dims = tile dims, dim0 stride}.
// ---------------------------------------------------------------------------
static __device__ __forceinline__ void tdm_load_2d(unsigned lds_byte_addr,
                                                   const u16* gsrc, int rows,
                                                   int stride_elems,
                                                   unsigned pad_int,
                                                   unsigned pad_amt,
                                                   int w_elems) {
  const u64 ga = (u64)gsrc;
  u32x4 g0;
  g0[0] = 1u;                                   // count=1 (valid descriptor)
  g0[1] = lds_byte_addr;                        // lds_addr
  g0[2] = (unsigned)ga;                         // global_addr[31:0]
  g0[3] = ((unsigned)(ga >> 32) & 0x01FFFFFFu)  // global_addr[56:32]
          | (2u << 30);                         // type=2 ("image")
  u32x8 g1;
  g1[0] = (1u << 16)                            // data_size: 2 bytes
          | (1u << 20)                          // pad_enable
          | (pad_int << 22) | (pad_amt << 25);
  g1[1] = ((unsigned)w_elems & 0xFFFFu) << 16;                    // tensor_dim0 lo16
  g1[2] = ((unsigned)w_elems >> 16) | (((unsigned)rows) << 16);   // dim0 hi | tensor_dim1 lo16
  g1[3] = ((unsigned)rows >> 16) | (((unsigned)w_elems) << 16);   // dim1 hi | tile_dim0
  g1[4] = (unsigned)rows;                                         // tile_dim1 (tile_dim2=0)
  g1[5] = (unsigned)stride_elems;                                 // tensor_dim0_stride lo32
  g1[6] = 0u;
  g1[7] = 0u;
  u32x4 gz;
  gz[0] = 0u; gz[1] = 0u; gz[2] = 0u; gz[3] = 0u;
  asm volatile("tensor_load_to_lds %0, %1, %2, %3"
               :: "s"(g0), "s"(g1), "s"(gz), "s"(gz)
               : "memory");
}

static __device__ __forceinline__ void tdm_wait() {
  __builtin_amdgcn_s_wait_tensorcnt(0);
}

// ---------------------------------------------------------------------------
// Step 1: circulant row  c[d] = (1/T) sum_k exp(-|a|*|f_k|) cos(2*pi*k*d/T)
// ---------------------------------------------------------------------------
__global__ void k_spectral_row(const float* __restrict__ alpha_p, float* __restrict__ c) {
  const int d = blockIdx.x * 256 + threadIdx.x;         // 0..1023
  const float alpha = fabsf(alpha_p[0]);
  float sum = 0.f;
  for (int k = 0; k < 1024; ++k) {
    int kk = (k <= 512) ? k : (1024 - k);               // |fftfreq|*T
    float w = __expf(-alpha * ((float)kk * (1.0f / 1024.0f)));
    int ph = (k * d) & 1023;                            // angle mod 2*pi
    sum += w * __cosf((float)ph * (6.2831853071795864f / 1024.0f));
  }
  c[d] = sum * (1.0f / 1024.0f);
}

__global__ void k_fill_cmat(const float* __restrict__ c, u16* __restrict__ Cm) {
  const int idx = blockIdx.x * 256 + threadIdx.x;       // t*1024 + s
  const int t = idx >> 10, s = idx & 1023;
  Cm[idx] = f2bf(c[(t - s) & 1023]);
}

__global__ void k_cvt_bf16(const float* __restrict__ src, u16* __restrict__ dst) {
  const int i = blockIdx.x * 256 + threadIdx.x;
  dst[i] = f2bf(src[i]);
}

// W [K=1024, N=1024] f32 -> WT [N, K] bf16 (writes coalesced, reads L2)
__global__ void k_transpose_w(const float* __restrict__ W, u16* __restrict__ WT) {
  const int idx = blockIdx.x * 256 + threadIdx.x;       // n*1024 + k
  const int n = idx >> 10, k = idx & 1023;
  WT[idx] = f2bf(W[k * 1024 + n]);
}

// per-batch bf16 transpose [T,D] -> [D,T] (grid.z = batch)
__global__ void k_transpose_b(const u16* __restrict__ X, u16* __restrict__ XT) {
  const int idx = blockIdx.x * 256 + threadIdx.x;       // n*1024 + k within batch
  const int n = idx >> 10, k = idx & 1023;
  const i64 base = (i64)blockIdx.z * 1024 * 1024;
  XT[base + idx] = X[base + (i64)k * 1024 + n];
}

// ---------------------------------------------------------------------------
// Tiled bf16 WMMA GEMM:  C[M,N] = A[M,K] * BT^T + bias, BT given [N,K].
// A and B tiles staged by per-wave TDM slices, double-buffered on TENSORcnt,
// one barrier per K-step. Block = 128 threads = 4 waves (2x2), block tile
// 64(M) x 128(N), K-step 32. LDS rows: 32 data + 8 pad elems (80B stride,
// TDM pad 16dw interval / 4dw amount). Grid: x = N/128, y = M/64, z = batch.
// ---------------------------------------------------------------------------
template <bool OUT_F32>
__global__ __launch_bounds__(128) void k_gemm(
    const u16* __restrict__ A, int lda, i64 sA,
    const u16* __restrict__ B, int ldb, i64 sB,
    void* __restrict__ Cv, int ldc, i64 sC,
    const float* __restrict__ bias, int K) {
  constexpr int LDS_S = 40;                              // 32 + 8 pad (16B-aligned rows)
  __shared__ u16 Al[2][64 * LDS_S];
  __shared__ u16 Bl[2][128 * LDS_S];

  const int tid = threadIdx.x;
  const int lane = tid & 31, w = tid >> 5;
  const int half = lane >> 4, lx = lane & 15;
  const int wm = (w >> 1) * 32, wn = (w & 1) * 64;
  const int m0 = blockIdx.y * 64, n0 = blockIdx.x * 128;

  A += (i64)blockIdx.z * sA;
  B += (i64)blockIdx.z * sB;
  float* Cf = (float*)Cv + (i64)blockIdx.z * sC;
  u16*   Cb = (u16*)Cv + (i64)blockIdx.z * sC;

  v8f acc[2][4];
#pragma unroll
  for (int mt = 0; mt < 2; ++mt)
#pragma unroll
    for (int nt = 0; nt < 4; ++nt) acc[mt][nt] = vzero8();

  // per-wave TDM slices: A rows [w*16, w*16+16), B rows [w*32, w*32+32)
  auto stage = [&](int buf, int k0) {
    tdm_load_2d((unsigned)(u64)&Al[buf][(w * 16) * LDS_S],
                A + (i64)(m0 + w * 16) * lda + k0, 16, lda, 3u, 3u, 32);
    tdm_load_2d((unsigned)(u64)&Bl[buf][(w * 32) * LDS_S],
                B + (i64)(n0 + w * 32) * ldb + k0, 32, ldb, 3u, 3u, 32);
  };

  stage(0, 0);

  for (int k0 = 0, it = 0; k0 < K; k0 += 32, ++it) {
    const int buf = it & 1;
    tdm_wait();                         // my TDM slices for tile `it` done
    __syncthreads();                    // everyone's slices visible
    if (k0 + 32 < K)                    // overlap next tile's DMA with compute
      stage(buf ^ 1, k0 + 32);

    v16bf af[2], bfrag[4];
#pragma unroll
    for (int mt = 0; mt < 2; ++mt)
      af[mt] = load_frag(&Al[buf][(wm + mt * 16 + lx) * LDS_S], 0, half);
#pragma unroll
    for (int nt = 0; nt < 4; ++nt)
      bfrag[nt] = load_frag(&Bl[buf][(wn + nt * 16 + lx) * LDS_S], 0, half);
#pragma unroll
    for (int mt = 0; mt < 2; ++mt)
#pragma unroll
      for (int nt = 0; nt < 4; ++nt)
        acc[mt][nt] = wmma_bf16(af[mt], bfrag[nt], acc[mt][nt]);
  }

  // Epilogue: C layout = VGPR i -> row (mt*16 + i + half*8), lane lx -> col.
#pragma unroll
  for (int mt = 0; mt < 2; ++mt)
#pragma unroll
    for (int nt = 0; nt < 4; ++nt)
#pragma unroll
      for (int i = 0; i < 8; ++i) {
        const int m = m0 + wm + mt * 16 + i + half * 8;
        const int n = n0 + wn + nt * 16 + lx;
        float v = acc[mt][nt][i];
        if (bias) v += bias[n];
        if constexpr (OUT_F32) Cf[(i64)m * ldc + n] = v;
        else                   Cb[(i64)m * ldc + n] = f2bf(v);
      }
}

// ---------------------------------------------------------------------------
// Flash attention: per (b, h, 128 q-rows) block, 4 waves x 32 q-rows each,
// streaming 64-key chunks. S = Q'K'^T * 1/8 + bias_scale*pi_bias, online
// softmax, O += P*V.  K chunks staged via TDM; V transposed on LDS store;
// pi_bias pre-converted to bf16.
// ---------------------------------------------------------------------------
__global__ __launch_bounds__(128) void k_flash(
    const u16* __restrict__ Qf, const u16* __restrict__ Kf,
    const u16* __restrict__ Vb, const u16* __restrict__ pib,
    const float* __restrict__ bias_scale_p, u16* __restrict__ Out) {
  constexpr int T = 1024, D = 1024, HD = 64, LDS_S = 72;  // 72*2B rows, 16B aligned
  __shared__ u16 Kl[64 * LDS_S];   // [key][hd]
  __shared__ u16 Vt[64 * LDS_S];   // [hd][key]
  __shared__ u16 Pl[4 * 32 * LDS_S];

  const int tid = threadIdx.x;
  const int lane = tid & 31, w = tid >> 5;
  const int half = lane >> 4, lx = lane & 15;
  const int b = blockIdx.z, h = blockIdx.y;
  const int wq = blockIdx.x * 128 + w * 32;               // this wave's first q row
  const float bscale = bias_scale_p[0];
  const float scale = 0.125f;                             // 1/sqrt(64)
  u16* Pw = Pl + w * 32 * LDS_S;

  // Q fragments for this wave: [mt][kk] over hd 0..63
  v16bf aq[2][2];
#pragma unroll
  for (int mt = 0; mt < 2; ++mt) {
    const u16* row = Qf + (i64)(b * T + wq + mt * 16 + lx) * D + h * HD;
#pragma unroll
    for (int kk = 0; kk < 2; ++kk) aq[mt][kk] = load_frag(row, kk * 32, half);
  }

  v8f oacc[2][4];
  float mst[2][8], lst[2][8];
#pragma unroll
  for (int mt = 0; mt < 2; ++mt) {
#pragma unroll
    for (int ht = 0; ht < 4; ++ht) oacc[mt][ht] = vzero8();
#pragma unroll
    for (int i = 0; i < 8; ++i) { mst[mt][i] = -3e38f; lst[mt][i] = 0.f; }
  }

  for (int j = 0; j < 16; ++j) {
    const int s0 = j * 64;
    // stage K chunk [64 keys][64 hd] via TDM (16 keys per wave, 128B+16B pad)
    tdm_load_2d((unsigned)(u64)&Kl[(w * 16) * LDS_S],
                Kf + (i64)(b * T + s0 + w * 16) * D + h * HD, 16, D, 4u, 3u, 64);
    // stage V chunk transposed: Vt[hd][key]
    {
      const int r = tid >> 1, hs = tid & 1;
      const u16* src = Vb + (i64)(b * T + s0 + r) * D + h * HD + hs * 32;
#pragma unroll 8
      for (int i = 0; i < 32; ++i) Vt[(hs * 32 + i) * LDS_S + r] = src[i];
    }
    tdm_wait();
    __syncthreads();

    // S = Q K^T (keys across lanes as N)
    v8f sacc[2][4];
#pragma unroll
    for (int mt = 0; mt < 2; ++mt)
#pragma unroll
      for (int nt = 0; nt < 4; ++nt) sacc[mt][nt] = vzero8();
#pragma unroll
    for (int kk = 0; kk < 2; ++kk) {
      v16bf bk[4];
#pragma unroll
      for (int nt = 0; nt < 4; ++nt)
        bk[nt] = load_frag(&Kl[(nt * 16 + lx) * LDS_S], kk * 32, half);
#pragma unroll
      for (int mt = 0; mt < 2; ++mt)
#pragma unroll
        for (int nt = 0; nt < 4; ++nt)
          sacc[mt][nt] = wmma_bf16(aq[mt][kk], bk[nt], sacc[mt][nt]);
    }

    // scale + pi bias (bf16)
#pragma unroll
    for (int mt = 0; mt < 2; ++mt)
#pragma unroll
      for (int nt = 0; nt < 4; ++nt)
#pragma unroll
        for (int i = 0; i < 8; ++i) {
          const int tg = wq + mt * 16 + i + half * 8;
          const int sg = s0 + nt * 16 + lx;
          sacc[mt][nt][i] = sacc[mt][nt][i] * scale +
                            bscale * bf2f(pib[(i64)(h * T + tg) * T + sg]);
        }

    // online softmax (rows in VGPR components, N across 16-lane halves)
#pragma unroll
    for (int mt = 0; mt < 2; ++mt)
#pragma unroll
      for (int i = 0; i < 8; ++i) {
        float rm = fmaxf(fmaxf(sacc[mt][0][i], sacc[mt][1][i]),
                         fmaxf(sacc[mt][2][i], sacc[mt][3][i]));
#pragma unroll
        for (int msk = 1; msk < 16; msk <<= 1) rm = fmaxf(rm, __shfl_xor(rm, msk, 32));
        const float mn = fmaxf(mst[mt][i], rm);
        const float al = __expf(mst[mt][i] - mn);
        float rs = 0.f;
#pragma unroll
        for (int nt = 0; nt < 4; ++nt) {
          float p = __expf(sacc[mt][nt][i] - mn);
          sacc[mt][nt][i] = p;
          rs += p;
        }
#pragma unroll
        for (int msk = 1; msk < 16; msk <<= 1) rs += __shfl_xor(rs, msk, 32);
        lst[mt][i] = lst[mt][i] * al + rs;
        mst[mt][i] = mn;
#pragma unroll
        for (int ht = 0; ht < 4; ++ht) oacc[mt][ht][i] *= al;
      }

    // P (C layout) -> per-wave LDS -> A layout
#pragma unroll
    for (int mt = 0; mt < 2; ++mt)
#pragma unroll
      for (int nt = 0; nt < 4; ++nt)
#pragma unroll
        for (int i = 0; i < 8; ++i)
          Pw[(mt * 16 + i + half * 8) * LDS_S + nt * 16 + lx] = f2bf(sacc[mt][nt][i]);

    // O += P * V   (hd across lanes as N; keys are the K dim)
#pragma unroll
    for (int kk = 0; kk < 2; ++kk) {
      v16bf ap[2], bv[4];
#pragma unroll
      for (int mt = 0; mt < 2; ++mt)
        ap[mt] = load_frag(Pw + (mt * 16 + lx) * LDS_S, kk * 32, half);
#pragma unroll
      for (int ht = 0; ht < 4; ++ht)
        bv[ht] = load_frag(&Vt[(ht * 16 + lx) * LDS_S], kk * 32, half);
#pragma unroll
      for (int mt = 0; mt < 2; ++mt)
#pragma unroll
        for (int ht = 0; ht < 4; ++ht)
          oacc[mt][ht] = wmma_bf16(ap[mt], bv[ht], oacc[mt][ht]);
    }
    __syncthreads();
  }

  // normalize + write attn output (bf16, heads interleaved in D)
#pragma unroll
  for (int mt = 0; mt < 2; ++mt)
#pragma unroll
    for (int ht = 0; ht < 4; ++ht)
#pragma unroll
      for (int i = 0; i < 8; ++i) {
        const int tg = wq + mt * 16 + i + half * 8;
        const int cg = h * HD + ht * 16 + lx;
        Out[(i64)(b * T + tg) * D + cg] = f2bf(oacc[mt][ht][i] / lst[mt][i]);
      }
}

// ---------------------------------------------------------------------------
extern "C" void kernel_launch(void* const* d_in, const int* in_sizes, int n_in,
                              void* d_out, int out_size, void* d_ws, size_t ws_size,
                              hipStream_t stream) {
  const float* x   = (const float*)d_in[0];
  const float* Wq  = (const float*)d_in[1];
  const float* bq  = (const float*)d_in[2];
  const float* Wk  = (const float*)d_in[3];
  const float* bk  = (const float*)d_in[4];
  const float* Wv  = (const float*)d_in[5];
  const float* bv  = (const float*)d_in[6];
  const float* Wo  = (const float*)d_in[7];
  const float* bo  = (const float*)d_in[8];
  const float* pib = (const float*)d_in[9];
  const float* bsc = (const float*)d_in[10];
  const float* alp = (const float*)d_in[11];
  float* out = (float*)d_out;

  char* ws = (char*)d_ws;
  size_t off = 0;
  auto alloc = [&](size_t bytes) -> void* {
    void* p = ws + off;
    off += (bytes + 255) & ~(size_t)255;
    return p;
  };
  const size_t SZ_DD  = (size_t)1024 * 1024 * sizeof(u16);   // 2 MiB
  const size_t SZ_BTD = (size_t)4096 * 1024 * sizeof(u16);   // 8 MiB

  float* cvec = (float*)alloc(1024 * sizeof(float));
  u16* Cmat = (u16*)alloc(SZ_DD);
  u16* xb   = (u16*)alloc(SZ_BTD);      // also reused as QbT after projections
  u16* WqT  = (u16*)alloc(SZ_DD);
  u16* WkT  = (u16*)alloc(SZ_DD);
  u16* WvT  = (u16*)alloc(SZ_DD);
  u16* WoT  = (u16*)alloc(SZ_DD);
  u16* Qb   = (u16*)alloc(SZ_BTD);
  u16* Kb   = (u16*)alloc(SZ_BTD);
  u16* Vb   = (u16*)alloc(SZ_BTD);
  u16* Qf   = (u16*)alloc(SZ_BTD);
  u16* Kf   = (u16*)alloc(SZ_BTD);
  u16* attn = (u16*)alloc(SZ_BTD);
  u16* KbT  = (u16*)alloc(SZ_BTD);
  u16* pibb = (u16*)alloc((size_t)16 * 1024 * 1024 * sizeof(u16));  // 32 MiB

  // 1) circulant spectral operator
  k_spectral_row<<<4, 256, 0, stream>>>(alp, cvec);
  k_fill_cmat<<<4096, 256, 0, stream>>>(cvec, Cmat);

  // 2) precision conversion / weight transposes / bf16 pi_bias
  k_cvt_bf16<<<16384, 256, 0, stream>>>(x, xb);
  k_cvt_bf16<<<65536, 256, 0, stream>>>(pib, pibb);
  k_transpose_w<<<4096, 256, 0, stream>>>(Wq, WqT);
  k_transpose_w<<<4096, 256, 0, stream>>>(Wk, WkT);
  k_transpose_w<<<4096, 256, 0, stream>>>(Wv, WvT);
  k_transpose_w<<<4096, 256, 0, stream>>>(Wo, WoT);

  // 3) Q/K/V projections: [4096,1024] x [1024,1024] + bias
  dim3 gproj(1024 / 128, 4096 / 64, 1);
  k_gemm<false><<<gproj, 128, 0, stream>>>(xb, 1024, 0, WqT, 1024, 0,
                                           (void*)Qb, 1024, 0, bq, 1024);
  k_gemm<false><<<gproj, 128, 0, stream>>>(xb, 1024, 0, WkT, 1024, 0,
                                           (void*)Kb, 1024, 0, bk, 1024);
  k_gemm<false><<<gproj, 128, 0, stream>>>(xb, 1024, 0, WvT, 1024, 0,
                                           (void*)Vb, 1024, 0, bv, 1024);

  // 4) spectral filter as batched GEMM: Q'[b] = Cmat @ Q[b].
  //    Pre-transpose Q/K per batch so the TDM-staged [N,K] path applies.
  //    (xb is dead after step 3 -> reuse it for QbT.)
  u16* QbT = xb;
  dim3 gtr(4096, 1, 4);
  k_transpose_b<<<gtr, 256, 0, stream>>>(Qb, QbT);
  k_transpose_b<<<gtr, 256, 0, stream>>>(Kb, KbT);
  const i64 sBT = (i64)1024 * 1024;
  dim3 gspec(1024 / 128, 1024 / 64, 4);
  k_gemm<false><<<gspec, 128, 0, stream>>>(Cmat, 1024, 0, QbT, 1024, sBT,
                                           (void*)Qf, 1024, sBT, nullptr, 1024);
  k_gemm<false><<<gspec, 128, 0, stream>>>(Cmat, 1024, 0, KbT, 1024, sBT,
                                           (void*)Kf, 1024, sBT, nullptr, 1024);

  // 5) flash attention with pi bias
  dim3 gfl(1024 / 128, 16, 4);
  k_flash<<<gfl, 128, 0, stream>>>(Qf, Kf, Vb, pibb, bsc, attn);

  // 6) output projection -> f32 d_out
  k_gemm<true><<<gproj, 128, 0, stream>>>(attn, 1024, 0, WoT, 1024, 0,
                                          (void*)out, 1024, 0, bo, 1024);
}
